// MultiHead_attention_61151744360696
// MI455X (gfx1250) — compile-verified, matrix operand order checked by reading, not run on hardware
//
#include <hip/hip_runtime.h>

// ---------------------------------------------------------------------------
// MHA for gfx1250 (MI455X): bf16 WMMA, flash attention (64-key blocks),
// fragment-packed intermediates so every hot-loop operand is one coalesced
// v16bf (2 x global_load_b128) load.  GEMMs register-blocked 32x64.
// ---------------------------------------------------------------------------

typedef __bf16 bf16;
typedef __attribute__((ext_vector_type(16))) __bf16 v16bf;
typedef __attribute__((ext_vector_type(2)))  __bf16 bf16x2;
typedef __attribute__((ext_vector_type(8)))  float  v8f;
typedef __attribute__((ext_vector_type(2)))  float  f32x2;

#define D_MODEL 512
#define NH 8
#define DH 64
#define BB 2
#define SS 2048
#define MROWS (BB * SS)       // 4096
#define FRAG_ELEMS 512        // one 16x32 bf16 fragment = 32 lanes * 16 elems

// Forward map: VGPR r of lane L holds K = kbase..kbase+1,
//   kbase = (r>=4 ? 16 : 0) + (L>=16 ? 8 : 0) + (r&3)*2
__device__ __forceinline__ int frag_kbase(int r, int half) {
  return ((r >> 2) << 4) | (half << 3) | ((r & 3) << 1);
}
// Inverse map: element (rc = row/col 0..15, kl = k 0..31) lives at
// lane = rc | ((kl&8)<<1), elem index = ((kl&16)>>1) | (kl&7).
__device__ __forceinline__ int frag_lane(int rc, int kl) {
  return rc | ((kl & 8) << 1);
}
__device__ __forceinline__ int frag_elem(int kl) {
  return ((kl & 16) >> 1) | (kl & 7);
}

__device__ __forceinline__ v8f wmma_bf16(v16bf a, v16bf b, v8f c) {
  return __builtin_amdgcn_wmma_f32_16x16x32_bf16(
      false, a, false, b, (short)0, c, false, false);
}

// ---------------------------------------------------------------------------
// Kernel 0: convert Wq / Wc (fp32 [n][k]) into bf16 B-fragment layout:
//   Wp[nt (32)][kk (16)] -> one 16x32 fragment (512 bf16).
// ---------------------------------------------------------------------------
__global__ __launch_bounds__(128)
void pack_weights_kernel(const float* __restrict__ Wq, const float* __restrict__ Wc,
                         bf16* __restrict__ Wqp, bf16* __restrict__ Wcp) {
  const int lane = threadIdx.x & 31;
  const int wave = threadIdx.x >> 5;
  const int half = lane >> 4;
  const int lcol = lane & 15;

  int task = blockIdx.x * 4 + wave;   // 0..1023
  int w    = task >> 9;               // 0 = Wq, 1 = Wc
  int rem  = task & 511;
  int nt   = rem >> 4;                // 0..31
  int kk   = rem & 15;                // 0..15

  const float* W = w ? Wc : Wq;
  bf16*        P = w ? Wcp : Wqp;

  const int col = nt * 16 + lcol;
  v16bf frag;
#pragma unroll
  for (int r = 0; r < 8; ++r) {
    int kb = kk * 32 + frag_kbase(r, half);
    f32x2 x = *(const f32x2*)(W + (size_t)col * D_MODEL + kb);
    frag[2 * r] = (bf16)x.x; frag[2 * r + 1] = (bf16)x.y;
  }
  *(v16bf*)(P + ((size_t)(nt * 16 + kk)) * FRAG_ELEMS + lane * 16) = frag;
}

// ---------------------------------------------------------------------------
// Kernel 1: fused QKV projection, 32x64 tile per wave (8 WMMAs per K-step,
// 2 A-frags x 4 B-frags).  Outputs scattered into fragment-packed layouts:
//   Qp[bh][qt (128)][kk (2)]  A-fragments (Q pre-scaled by 1/sqrt(dh))
//   Kp[bh][kt (128)][kk (2)]  B-fragments for Q*K^T
//   Vp[bh][kblk (64)][dt (4)] B-fragments for P*V
// ---------------------------------------------------------------------------
__global__ __launch_bounds__(128)
void proj_qkv_kernel(const float* __restrict__ q, const float* __restrict__ k,
                     const float* __restrict__ v, const bf16* __restrict__ Wqp,
                     const float* __restrict__ bias,
                     bf16* __restrict__ Qp, bf16* __restrict__ Kp,
                     bf16* __restrict__ Vp) {
  const int lane = threadIdx.x & 31;
  const int wave = threadIdx.x >> 5;
  const int half = lane >> 4;
  const int lcol = lane & 15;
  const int lbase = lane * 16;

  const int MT2 = MROWS / 32;         // 128
  const int NG  = D_MODEL / 64;       // 8
  int task = blockIdx.x * 4 + wave;   // 0..3071
  int mat  = task / (MT2 * NG);
  int rem  = task - mat * (MT2 * NG);
  int mt2  = rem / NG;
  int ng   = rem - mt2 * NG;
  int m0 = mt2 * 32, n0 = ng * 64;

  const float* X = (mat == 0) ? q : (mat == 1) ? k : v;

  v8f acc[2][4];
#pragma unroll
  for (int mi = 0; mi < 2; ++mi)
#pragma unroll
    for (int t = 0; t < 4; ++t) acc[mi][t] = (v8f){};

  for (int kk = 0; kk < D_MODEL / 32; ++kk) {
    v16bf a[2];
#pragma unroll
    for (int mi = 0; mi < 2; ++mi) {
      const int arow = m0 + mi * 16 + lcol;
#pragma unroll
      for (int r = 0; r < 8; ++r) {
        int kb = kk * 32 + frag_kbase(r, half);
        f32x2 x = *(const f32x2*)(X + (size_t)arow * D_MODEL + kb);
        a[mi][2 * r] = (bf16)x.x; a[mi][2 * r + 1] = (bf16)x.y;
      }
    }
#pragma unroll
    for (int t = 0; t < 4; ++t) {
      int nt = (n0 >> 4) + t;
      v16bf b = *(const v16bf*)(Wqp + ((size_t)(nt * 16 + kk)) * FRAG_ELEMS + lbase);
      acc[0][t] = wmma_bf16(a[0], b, acc[0][t]);
      acc[1][t] = wmma_bf16(a[1], b, acc[1][t]);
    }
  }

#pragma unroll
  for (int t = 0; t < 4; ++t) {
    int n = n0 + t * 16 + lcol;
    float bias_v = bias[n];
    int h = n >> 6, d = n & 63;
    int bh_off = d;  // d used below
#pragma unroll
    for (int mi = 0; mi < 2; ++mi) {
#pragma unroll
      for (int r = 0; r < 8; ++r) {
        int m  = m0 + mi * 16 + r + (half << 3);
        int bi = m >> 11;
        int s  = m & (SS - 1);
        int bh = bi * NH + h;
        float val = acc[mi][t][r] + bias_v;
        if (mat == 0) {
          int qt = s >> 4, rc = s & 15, kk2 = d >> 5, kl = d & 31;
          Qp[((size_t)((bh * 128 + qt) * 2 + kk2)) * FRAG_ELEMS +
             frag_lane(rc, kl) * 16 + frag_elem(kl)] = (bf16)(val * 0.125f);
        } else if (mat == 1) {
          int kt = s >> 4, rc = s & 15, kk2 = d >> 5, kl = d & 31;
          Kp[((size_t)((bh * 128 + kt) * 2 + kk2)) * FRAG_ELEMS +
             frag_lane(rc, kl) * 16 + frag_elem(kl)] = (bf16)val;
        } else {
          int kblk = s >> 5, kl = s & 31, dt = d >> 4, c = d & 15;
          Vp[((size_t)((bh * 64 + kblk) * 4 + dt)) * FRAG_ELEMS +
             frag_lane(c, kl) * 16 + frag_elem(kl)] = (bf16)val;
        }
      }
    }
    (void)bh_off;
  }
}

// ---------------------------------------------------------------------------
// Kernel 2: flash attention, 64-key blocks.  One wave = 16 query rows of one
// (b,h); 16 WMMAs per iteration against one fixed-cost softmax reduction.
// Output scattered into A-fragment layout Op[mt (256)][kk (16)].
// ---------------------------------------------------------------------------
__global__ __launch_bounds__(128)
void attention_kernel(const bf16* __restrict__ Qp, const bf16* __restrict__ Kp,
                      const bf16* __restrict__ Vp, bf16* __restrict__ Op) {
  __shared__ bf16 smem[4][16][64];   // per-wave 16x64 P transpose buffer

  const int lane = threadIdx.x & 31;
  const int wave = threadIdx.x >> 5;
  const int half = lane >> 4;
  const int lcol = lane & 15;
  const int lbase = lane * 16;

  const int QT = SS / 16;             // 128
  int task = blockIdx.x * 4 + wave;
  int bh = task / QT;
  int qt = task - bh * QT;
  int q0 = qt * 16;

  const bf16* Kpb = Kp + (size_t)bh * 128 * 2 * FRAG_ELEMS;
  const bf16* Vpb = Vp + (size_t)bh * 64 * 4 * FRAG_ELEMS;

  v16bf aq[2];
  {
    const bf16* Qpb = Qp + ((size_t)(bh * 128 + qt) * 2) * FRAG_ELEMS;
    aq[0] = *(const v16bf*)(Qpb + lbase);
    aq[1] = *(const v16bf*)(Qpb + FRAG_ELEMS + lbase);
  }

  float m_i[8], l_i[8];
  v8f accO[4];
#pragma unroll
  for (int r = 0; r < 8; ++r) { m_i[r] = -1e30f; l_i[r] = 0.f; }
#pragma unroll
  for (int t = 0; t < 4; ++t) accO[t] = (v8f){};

  for (int kb64 = 0; kb64 < SS / 64; ++kb64) {   // 32 iterations
    // --- scores: four 16x16 tiles over 64 keys (8 WMMAs) ---
    v8f sc[4] = {(v8f){}, (v8f){}, (v8f){}, (v8f){}};
#pragma unroll
    for (int t = 0; t < 4; ++t) {
      const bf16* kf = Kpb + ((size_t)((kb64 * 4 + t) * 2)) * FRAG_ELEMS + lbase;
#pragma unroll
      for (int kk = 0; kk < 2; ++kk) {
        v16bf bK = *(const v16bf*)(kf + kk * FRAG_ELEMS);
        sc[t] = wmma_bf16(aq[kk], bK, sc[t]);
      }
    }

    // --- online softmax (rows live per 16-lane half) ---
    float mx[8];
#pragma unroll
    for (int r = 0; r < 8; ++r)
      mx[r] = fmaxf(fmaxf(sc[0][r], sc[1][r]), fmaxf(sc[2][r], sc[3][r]));
#pragma unroll
    for (int off = 1; off < 16; off <<= 1)
#pragma unroll
      for (int r = 0; r < 8; ++r)
        mx[r] = fmaxf(mx[r], __shfl_xor(mx[r], off, 32));

    float alpha[8], rs[8];
#pragma unroll
    for (int r = 0; r < 8; ++r) {
      float mn = fmaxf(m_i[r], mx[r]);
      alpha[r] = __expf(m_i[r] - mn);
      m_i[r] = mn;
      float p0 = __expf(sc[0][r] - mn);
      float p1 = __expf(sc[1][r] - mn);
      float p2 = __expf(sc[2][r] - mn);
      float p3 = __expf(sc[3][r] - mn);
      sc[0][r] = p0; sc[1][r] = p1; sc[2][r] = p2; sc[3][r] = p3;
      rs[r] = (p0 + p1) + (p2 + p3);
    }
#pragma unroll
    for (int off = 1; off < 16; off <<= 1)
#pragma unroll
      for (int r = 0; r < 8; ++r)
        rs[r] += __shfl_xor(rs[r], off, 32);
#pragma unroll
    for (int r = 0; r < 8; ++r) l_i[r] = l_i[r] * alpha[r] + rs[r];
#pragma unroll
    for (int t = 0; t < 4; ++t)
#pragma unroll
      for (int r = 0; r < 8; ++r) accO[t][r] *= alpha[r];

    // --- transpose P (16x64) via LDS: D-layout out, A-layout in ---
#pragma unroll
    for (int t = 0; t < 4; ++t)
#pragma unroll
      for (int r = 0; r < 8; ++r)
        smem[wave][r + (half << 3)][t * 16 + lcol] = (bf16)sc[t][r];

    __asm volatile("s_wait_dscnt 0x0" ::: "memory");

    v16bf pa[2];
#pragma unroll
    for (int kk2 = 0; kk2 < 2; ++kk2)
#pragma unroll
      for (int r = 0; r < 8; ++r) {
        int kb = kk2 * 32 + frag_kbase(r, half);
        bf16x2 p = *(const bf16x2*)(&smem[wave][lcol][kb]);
        pa[kk2][2 * r] = p.x; pa[kk2][2 * r + 1] = p.y;
      }

    // --- accO += P(16x64) x V(64x64): 8 WMMAs ---
#pragma unroll
    for (int t = 0; t < 4; ++t) {
#pragma unroll
      for (int kk2 = 0; kk2 < 2; ++kk2) {
        v16bf bV = *(const v16bf*)(Vpb +
            ((size_t)((kb64 * 2 + kk2) * 4 + t)) * FRAG_ELEMS + lbase);
        accO[t] = wmma_bf16(pa[kk2], bV, accO[t]);
      }
    }
  }

  // --- normalize + scatter into A-fragment layout for output projection ---
  float inv[8];
#pragma unroll
  for (int r = 0; r < 8; ++r) inv[r] = 1.0f / l_i[r];

  const int bi = bh >> 3;
  const int h  = bh & 7;
#pragma unroll
  for (int t = 0; t < 4; ++t) {
    int c  = h * DH + t * 16 + lcol;    // column in [0,512)
    int kk = c >> 5, kl = c & 31;
    int le = frag_lane(0, kl) * 16 + frag_elem(kl);
#pragma unroll
    for (int r = 0; r < 8; ++r) {
      int m  = bi * SS + q0 + r + (half << 3);
      int mt = m >> 4, rc = m & 15;
      Op[((size_t)(mt * 16 + kk)) * FRAG_ELEMS + le + rc * 16] =
          (bf16)(accO[t][r] * inv[r]);
    }
  }
}

// ---------------------------------------------------------------------------
// Kernel 3: output projection Y = O @ Wc^T + b, 32x64 tile per wave.
// ---------------------------------------------------------------------------
__global__ __launch_bounds__(128)
void out_proj_kernel(const bf16* __restrict__ Op, const bf16* __restrict__ Wcp,
                     const float* __restrict__ bias, float* __restrict__ Y) {
  const int lane = threadIdx.x & 31;
  const int wave = threadIdx.x >> 5;
  const int half = lane >> 4;
  const int lcol = lane & 15;
  const int lbase = lane * 16;

  const int NG = D_MODEL / 64;        // 8
  int task = blockIdx.x * 4 + wave;   // 0..1023
  int mt2 = task / NG;                // 0..127 (32-row tiles)
  int ng  = task - mt2 * NG;
  int n0 = ng * 64;

  v8f acc[2][4];
#pragma unroll
  for (int mi = 0; mi < 2; ++mi)
#pragma unroll
    for (int t = 0; t < 4; ++t) acc[mi][t] = (v8f){};

  for (int kk = 0; kk < D_MODEL / 32; ++kk) {
    v16bf a[2];
#pragma unroll
    for (int mi = 0; mi < 2; ++mi)
      a[mi] = *(const v16bf*)(Op +
          ((size_t)((mt2 * 2 + mi) * 16 + kk)) * FRAG_ELEMS + lbase);
#pragma unroll
    for (int t = 0; t < 4; ++t) {
      int nt = (n0 >> 4) + t;
      v16bf b = *(const v16bf*)(Wcp + ((size_t)(nt * 16 + kk)) * FRAG_ELEMS + lbase);
      acc[0][t] = wmma_bf16(a[0], b, acc[0][t]);
      acc[1][t] = wmma_bf16(a[1], b, acc[1][t]);
    }
  }

#pragma unroll
  for (int t = 0; t < 4; ++t) {
    int n = n0 + t * 16 + lcol;
    float bias_v = bias[n];
#pragma unroll
    for (int mi = 0; mi < 2; ++mi)
#pragma unroll
      for (int r = 0; r < 8; ++r) {
        int m = mt2 * 32 + mi * 16 + r + (half << 3);
        Y[(size_t)m * D_MODEL + n] = acc[mi][t][r] + bias_v;
      }
  }
}

// ---------------------------------------------------------------------------
// Launch
// ---------------------------------------------------------------------------
extern "C" void kernel_launch(void* const* d_in, const int* in_sizes, int n_in,
                              void* d_out, int out_size, void* d_ws, size_t ws_size,
                              hipStream_t stream) {
  const float* q   = (const float*)d_in[0];
  const float* k   = (const float*)d_in[1];
  const float* v   = (const float*)d_in[2];
  const float* Wq  = (const float*)d_in[3];
  const float* Wqb = (const float*)d_in[4];
  const float* Wc  = (const float*)d_in[5];
  const float* Wcb = (const float*)d_in[6];

  // workspace: Wqp(512K) | Wcp(512K) | Qp(4M) | Kp(4M) | Vp(4M) | Op(4M) = 17MB
  char* ws = (char*)d_ws;
  bf16* Wqp = (bf16*)(ws);
  bf16* Wcp = (bf16*)(ws + (size_t)(512u << 10));
  bf16* Qp  = (bf16*)(ws + (size_t)(1u  << 20));
  bf16* Kp  = (bf16*)(ws + (size_t)(5u  << 20));
  bf16* Vp  = (bf16*)(ws + (size_t)(9u  << 20));
  bf16* Op  = (bf16*)(ws + (size_t)(13u << 20));

  // 0) pack weights into WMMA fragments (1024 waves)
  pack_weights_kernel<<<dim3(256), dim3(128), 0, stream>>>(Wq, Wc, Wqp, Wcp);

  // 1) QKV projection: 3 * 128 row-tiles(32) * 8 col-groups = 3072 waves
  proj_qkv_kernel<<<dim3(768), dim3(128), 0, stream>>>(
      q, k, v, Wqp, Wqb, Qp, Kp, Vp);

  // 2) flash attention: 16 (b,h) * 128 q-tiles = 2048 waves
  attention_kernel<<<dim3(512), dim3(128), 0, stream>>>(Qp, Kp, Vp, Op);

  // 3) output projection: 128 * 8 = 1024 waves
  out_proj_kernel<<<dim3(256), dim3(128), 0, stream>>>(Op, Wcp, Wcb, (float*)d_out);
}
